// _MultiHeadAttention_65687229825624
// MI455X (gfx1250) — compile-verified, matrix operand order checked by reading, not run on hardware
//
#include <hip/hip_runtime.h>
#include <hip/hip_bf16.h>

#define B_SZ 4
#define T_SZ 2048
#define D_SZ 1024
#define H_SZ 16
#define DH_SZ 64

typedef __attribute__((ext_vector_type(16))) _Float16 v16h;
typedef __attribute__((ext_vector_type(8)))  float    v8f;

static __device__ __forceinline__ v8f wmma16x16x32(v16h a, v16h b, v8f c) {
  // D = A(16x32 f16) * B(32x16 f16) + C(16x16 f32)
  return __builtin_amdgcn_wmma_f32_16x16x32_f16(false, a, false, b, (short)0, c,
                                                false, false);
}

// ---- WMMA operand loaders (CDNA5 16-bit 16x32 A layout) ----
// Lane l holds row m = l&15. For lanes 0-15: elems 0..7 -> K=k0+0..7,
// elems 8..15 -> K=k0+16..23. For lanes 16-31: +8 on both K groups.
// Also serves as B (32x16) loader when the stored matrix is B^T.
static __device__ __forceinline__ v16h load_tile_f32(const float* p, int ld,
                                                     int row0, int k0, int lane) {
  int m  = lane & 15;
  int kk = k0 + ((lane >> 4) << 3);
  const float* q = p + (size_t)(row0 + m) * ld + kk;
  v16h r;
#pragma unroll
  for (int i = 0; i < 8; ++i) r[i] = (_Float16)q[i];
#pragma unroll
  for (int i = 0; i < 8; ++i) r[8 + i] = (_Float16)q[16 + i];
  return r;
}

static __device__ __forceinline__ v16h load_tile_f16(const _Float16* p, int ld,
                                                     int row0, int k0, int lane) {
  int m  = lane & 15;
  int kk = k0 + ((lane >> 4) << 3);
  const _Float16* q = p + (size_t)(row0 + m) * ld + kk;
  v16h r;
#pragma unroll
  for (int i = 0; i < 8; ++i) r[i] = q[i];
#pragma unroll
  for (int i = 0; i < 8; ++i) r[8 + i] = q[16 + i];
  return r;
}

// ---- one-shot fp32 -> f16 weight conversion (4 elems/thread, vectorizes) ----
__global__ __launch_bounds__(256) void cvt_w_kernel(const float* __restrict__ src,
                                                    _Float16* __restrict__ dst,
                                                    int n) {
  int i = (blockIdx.x * 256 + threadIdx.x) * 4;
  if (i + 3 < n) {
#pragma unroll
    for (int j = 0; j < 4; ++j) dst[i + j] = (_Float16)src[i + j];
  }
}

// ---- Kernel 1: input projections  y = x @ W^T ----
// Writes any of: fp32 head-split [B,H,T,DH] (K/V final outputs),
// f16 head-split [B,H,T,DH] (Q / K shadows), f16 transposed [B,H,DH,T] (V^T).
__global__ __launch_bounds__(32) void proj_kernel(
    const float* __restrict__ x, const _Float16* __restrict__ w16,
    float* __restrict__ out_f32, _Float16* __restrict__ out_f16_hs,
    _Float16* __restrict__ out_f16_tr) {
  int lane = threadIdx.x;
  int row0 = blockIdx.x * 16;  // over B*T
  int col0 = blockIdx.y * 64;  // 4 x 16 output features per wave
  v8f c0 = {}, c1 = {}, c2 = {}, c3 = {};
  for (int k0 = 0; k0 < D_SZ; k0 += 32) {
    v16h a = load_tile_f32(x, D_SZ, row0, k0, lane);
    c0 = wmma16x16x32(a, load_tile_f16(w16, D_SZ, col0 +  0, k0, lane), c0);
    c1 = wmma16x16x32(a, load_tile_f16(w16, D_SZ, col0 + 16, k0, lane), c1);
    c2 = wmma16x16x32(a, load_tile_f16(w16, D_SZ, col0 + 32, k0, lane), c2);
    c3 = wmma16x16x32(a, load_tile_f16(w16, D_SZ, col0 + 48, k0, lane), c3);
  }
  int half = lane >> 4, lm = lane & 15;
#pragma unroll
  for (int j = 0; j < 4; ++j) {
    int o = col0 + j * 16 + lm;
    int h = o >> 6, dh = o & 63;
#pragma unroll
    for (int g = 0; g < 8; ++g) {
      int r  = row0 + g + half * 8;
      int bb = r >> 11, t = r & (T_SZ - 1);  // T == 2048
      float v = (j == 0 ? c0[g] : j == 1 ? c1[g] : j == 2 ? c2[g] : c3[g]);
      size_t hs = (((size_t)bb * H_SZ + h) * T_SZ + t) * DH_SZ + dh;
      if (out_f32)    out_f32[hs]    = v;
      if (out_f16_hs) out_f16_hs[hs] = (_Float16)v;
      if (out_f16_tr)
        out_f16_tr[(((size_t)bb * H_SZ + h) * DH_SZ + dh) * T_SZ + t] = (_Float16)v;
    }
  }
}

// ---- Kernel 2: causal flash attention ----
// 8 waves per block; wave w owns query tile blockIdx.y*8+w of head blockIdx.x.
// Each 32-key K/V block is staged into LDS once and shared by all 8 waves.
__global__ __launch_bounds__(256) void attn_kernel(
    const _Float16* __restrict__ qh,   // [B*H, T, DH] f16
    const _Float16* __restrict__ kh,   // [B*H, T, DH] f16
    const _Float16* __restrict__ vt,   // [B*H, DH, T] f16 (V transposed)
    _Float16* __restrict__ attn_out)   // [B, T, D]    f16
{
  __shared__ __align__(16) _Float16 lds_k [32 * 64];      // K block  [key][dh]
  __shared__ __align__(16) _Float16 lds_vt[64 * 32];      // V^T block [dh][key]
  __shared__ __align__(16) _Float16 lds_p [8 * 16 * 32];  // per-wave P tiles

  int tid  = threadIdx.x;
  int lane = tid & 31;
  int w    = tid >> 5;                 // wave id 0..7
  int half = lane >> 4, lm = lane & 15;
  int bh = blockIdx.x;                 // 0..B*H-1
  int qt = blockIdx.y * 8 + w;         // this wave's query tile
  const _Float16* qbase  = qh + (size_t)bh * T_SZ * DH_SZ;
  const _Float16* kbase  = kh + (size_t)bh * T_SZ * DH_SZ;
  const _Float16* vtbase = vt + (size_t)bh * DH_SZ * T_SZ;
  _Float16* pbuf = lds_p + w * (16 * 32);

  v16h aq0 = load_tile_f16(qbase, DH_SZ, qt * 16, 0, lane);
  v16h aq1 = load_tile_f16(qbase, DH_SZ, qt * 16, 32, lane);

  v8f o0 = {}, o1 = {}, o2 = {}, o3 = {};
  float M[8], L[8];
#pragma unroll
  for (int g = 0; g < 8; ++g) { M[g] = -1e30f; L[g] = 0.0f; }

  // key blocks needed by the highest query tile in this block
  int nkb = blockIdx.y * 4 + 4;        // (blockIdx.y*128+127)/32 + 1
  int qend = qt * 16 + 15;

  for (int kb = 0; kb < nkb; ++kb) {
    int kcol0 = kb * 32;
    __syncthreads();  // previous iteration's LDS reads complete
    {   // cooperative staging: one uint4 (8 f16) per thread per buffer
      int off = tid * 8;
      *(uint4*)&lds_k[off] =
          *(const uint4*)&kbase[(size_t)kcol0 * DH_SZ + off];
      int dh = off >> 5, ko = off & 31;
      *(uint4*)&lds_vt[off] =
          *(const uint4*)&vtbase[(size_t)dh * T_SZ + kcol0 + ko];
    }
    __syncthreads();  // staging visible to all waves

    if (kcol0 <= qend) {  // causal: this wave needs the block
      // S = Q K^T over this 32-key block (two 16x16 C tiles), from LDS
      v8f s0 = {}, s1 = {};
      s0 = wmma16x16x32(aq0, load_tile_f16(lds_k, DH_SZ,  0,  0, lane), s0);
      s0 = wmma16x16x32(aq1, load_tile_f16(lds_k, DH_SZ,  0, 32, lane), s0);
      s1 = wmma16x16x32(aq0, load_tile_f16(lds_k, DH_SZ, 16,  0, lane), s1);
      s1 = wmma16x16x32(aq1, load_tile_f16(lds_k, DH_SZ, 16, 32, lane), s1);

      const float scale = 0.125f;  // 1/sqrt(64)
#pragma unroll
      for (int g = 0; g < 8; ++g) {
        int qrow = qt * 16 + g + half * 8;
        float x0 = s0[g] * scale; if (kcol0 + lm > qrow)      x0 = -1e30f;
        float x1 = s1[g] * scale; if (kcol0 + 16 + lm > qrow) x1 = -1e30f;
        // row max across the 16 lanes of this half-wave
        float r = fmaxf(x0, x1);
        r = fmaxf(r, __shfl_xor(r, 1));
        r = fmaxf(r, __shfl_xor(r, 2));
        r = fmaxf(r, __shfl_xor(r, 4));
        r = fmaxf(r, __shfl_xor(r, 8));
        float mnew  = fmaxf(M[g], r);
        float alpha = __expf(M[g] - mnew);
        M[g] = mnew;
        float p0 = __expf(x0 - mnew);
        float p1 = __expf(x1 - mnew);
        float srow = p0 + p1;
        srow += __shfl_xor(srow, 1);
        srow += __shfl_xor(srow, 2);
        srow += __shfl_xor(srow, 4);
        srow += __shfl_xor(srow, 8);
        L[g] = L[g] * alpha + srow;
        o0[g] *= alpha; o1[g] *= alpha; o2[g] *= alpha; o3[g] *= alpha;
        // stash P (C layout) into this wave's LDS slice
        int m = g + half * 8;
        pbuf[m * 32 + lm]      = (_Float16)p0;
        pbuf[m * 32 + 16 + lm] = (_Float16)p1;
      }
      __asm__ volatile("s_wait_dscnt 0" ::: "memory");  // wave-local P RAW
      v16h ap = load_tile_f16(pbuf, 32, 0, 0, lane);    // P as 16x32 A matrix
      // O += P @ V_block, V^T rows contiguous along keys (ds_load_b128)
      o0 = wmma16x16x32(ap, load_tile_f16(lds_vt, 32,  0, 0, lane), o0);
      o1 = wmma16x16x32(ap, load_tile_f16(lds_vt, 32, 16, 0, lane), o1);
      o2 = wmma16x16x32(ap, load_tile_f16(lds_vt, 32, 32, 0, lane), o2);
      o3 = wmma16x16x32(ap, load_tile_f16(lds_vt, 32, 48, 0, lane), o3);
    }
  }

  int b = bh >> 4, h = bh & 15;  // H == 16
#pragma unroll
  for (int g = 0; g < 8; ++g) {
    float inv = 1.0f / L[g];
    int t = qt * 16 + g + half * 8;
    size_t ro = ((size_t)b * T_SZ + t) * D_SZ + (size_t)h * DH_SZ;
    attn_out[ro + lm]      = (_Float16)(o0[g] * inv);
    attn_out[ro + 16 + lm] = (_Float16)(o1[g] * inv);
    attn_out[ro + 32 + lm] = (_Float16)(o2[g] * inv);
    attn_out[ro + 48 + lm] = (_Float16)(o3[g] * inv);
  }
}

// ---- Kernel 3: output projection  out = attn @ wo^T (fp32 result) ----
__global__ __launch_bounds__(32) void oproj_kernel(
    const _Float16* __restrict__ x, const _Float16* __restrict__ w16,
    float* __restrict__ out) {
  int lane = threadIdx.x;
  int row0 = blockIdx.x * 16;
  int col0 = blockIdx.y * 64;
  v8f c0 = {}, c1 = {}, c2 = {}, c3 = {};
  for (int k0 = 0; k0 < D_SZ; k0 += 32) {
    v16h a = load_tile_f16(x, D_SZ, row0, k0, lane);
    c0 = wmma16x16x32(a, load_tile_f16(w16, D_SZ, col0 +  0, k0, lane), c0);
    c1 = wmma16x16x32(a, load_tile_f16(w16, D_SZ, col0 + 16, k0, lane), c1);
    c2 = wmma16x16x32(a, load_tile_f16(w16, D_SZ, col0 + 32, k0, lane), c2);
    c3 = wmma16x16x32(a, load_tile_f16(w16, D_SZ, col0 + 48, k0, lane), c3);
  }
  int half = lane >> 4, lm = lane & 15;
#pragma unroll
  for (int g = 0; g < 8; ++g) {
    size_t ro = (size_t)(row0 + g + half * 8) * D_SZ + col0 + lm;
    out[ro]      = c0[g];
    out[ro + 16] = c1[g];
    out[ro + 32] = c2[g];
    out[ro + 48] = c3[g];
  }
}

extern "C" void kernel_launch(void* const* d_in, const int* in_sizes, int n_in,
                              void* d_out, int out_size, void* d_ws, size_t ws_size,
                              hipStream_t stream) {
  (void)in_sizes; (void)n_in; (void)out_size; (void)ws_size;
  const float* query = (const float*)d_in[0];
  const float* key   = (const float*)d_in[1];
  const float* value = (const float*)d_in[2];
  const float* wq    = (const float*)d_in[3];
  const float* wk    = (const float*)d_in[4];
  const float* wv    = (const float*)d_in[5];
  const float* wo    = (const float*)d_in[6];
  // d_in[7] (attn_mask) is the causal mask; applied analytically — not read.

  float* out   = (float*)d_out;                          // [B,T,D]
  float* k_out = out   + (size_t)B_SZ * T_SZ * D_SZ;     // [B,H,T,DH]
  float* v_out = k_out + (size_t)B_SZ * H_SZ * T_SZ * DH_SZ;

  const size_t ACT = (size_t)B_SZ * T_SZ * D_SZ;  // 8M elems
  const size_t WN  = (size_t)D_SZ * D_SZ;         // 1M elems
  _Float16* q_ws    = (_Float16*)d_ws;      // [B,H,T,DH] f16
  _Float16* attn_ws = q_ws    + ACT;        // [B,T,D]    f16
  _Float16* k16_ws  = attn_ws + ACT;        // [B,H,T,DH] f16
  _Float16* vt16_ws = k16_ws  + ACT;        // [B,H,DH,T] f16
  _Float16* wq16    = vt16_ws + ACT;
  _Float16* wk16    = wq16 + WN;
  _Float16* wv16    = wk16 + WN;
  _Float16* wo16    = wv16 + WN;

  // One-shot weight conversions (1M elems each, 4/thread)
  int cvtgrid = (int)(WN / 4 / 256);
  cvt_w_kernel<<<cvtgrid, 256, 0, stream>>>(wq, wq16, (int)WN);
  cvt_w_kernel<<<cvtgrid, 256, 0, stream>>>(wk, wk16, (int)WN);
  cvt_w_kernel<<<cvtgrid, 256, 0, stream>>>(wv, wv16, (int)WN);
  cvt_w_kernel<<<cvtgrid, 256, 0, stream>>>(wo, wo16, (int)WN);

  dim3 gproj(B_SZ * T_SZ / 16, D_SZ / 64);
  proj_kernel<<<gproj, 32, 0, stream>>>(query, wq16, nullptr, q_ws,  nullptr);
  proj_kernel<<<gproj, 32, 0, stream>>>(key,   wk16, k_out, k16_ws,  nullptr);
  proj_kernel<<<gproj, 32, 0, stream>>>(value, wv16, v_out, nullptr, vt16_ws);

  dim3 gattn(B_SZ * H_SZ, T_SZ / 128);   // 8 query tiles per block
  attn_kernel<<<gattn, 256, 0, stream>>>(q_ws, k16_ws, vt16_ws, attn_ws);

  oproj_kernel<<<gproj, 32, 0, stream>>>(attn_ws, wo16, out);
}
